// TimeMoD_48215302865732
// MI455X (gfx1250) — compile-verified, matrix operand order checked by reading, not run on hardware
//
#include <hip/hip_runtime.h>
#include <hip/hip_bf16.h>

// ---------------- types ----------------
typedef _Float16 v8h  __attribute__((ext_vector_type(8)));
typedef _Float16 v16h __attribute__((ext_vector_type(16)));
typedef float    v8f  __attribute__((ext_vector_type(8)));
typedef int      v4i  __attribute__((ext_vector_type(4)));

#define AS_GLOBAL __attribute__((address_space(1)))
#define AS_LDS    __attribute__((address_space(3)))

__device__ __forceinline__ v16h cat8(v8h lo, v8h hi) {
    return __builtin_shufflevector(lo, hi, 0,1,2,3,4,5,6,7,8,9,10,11,12,13,14,15);
}

// Problem constants
#define BB   16
#define TT   16
#define CC   128
#define HH   32
#define WW   32
#define KK   8
#define CHW  (CC*HH*WW)      // 131072

// ---- CDNA5 async global->LDS path (guarded: falls back cleanly) ----
#if defined(__HIP_DEVICE_COMPILE__) && __has_builtin(__builtin_amdgcn_global_load_async_to_lds_b128)
#define HAS_ASYNC_LDS 1
#else
#define HAS_ASYNC_LDS 0
#endif

__device__ __forceinline__ void wait_async0() {
#if defined(__HIP_DEVICE_COMPILE__)
#if __has_builtin(__builtin_amdgcn_s_wait_asynccnt)
    __builtin_amdgcn_s_wait_asynccnt(0);
#else
    asm volatile("s_wait_asynccnt 0" ::: "memory");
#endif
#endif
}

// ---------------- kernel 1: router dot products ----------------
__global__ __launch_bounds__(256)
void router_kernel(const float* __restrict__ x, const float* __restrict__ wr,
                   const float* __restrict__ br, float* __restrict__ weights) {
    int bt = blockIdx.x;                          // 0..255 = b*16+t
    const float4* xp = (const float4*)(x + (size_t)bt * CHW);
    const float4* wp = (const float4*)wr;
    float s = 0.f;
    for (int i = threadIdx.x; i < CHW/4; i += 256) {
        float4 a = xp[i], b = wp[i];
        s += a.x*b.x + a.y*b.y + a.z*b.z + a.w*b.w;
    }
    __shared__ float red[256];
    red[threadIdx.x] = s;
    __syncthreads();
    for (int off = 128; off > 0; off >>= 1) {
        if (threadIdx.x < off) red[threadIdx.x] += red[threadIdx.x + off];
        __syncthreads();
    }
    if (threadIdx.x == 0) weights[bt] = red[0] + br[0];
}

// ---------------- kernel 2: top-K selection (rank with jax tie-break) -------
__global__ __launch_bounds__(256)
void select_kernel(const float* __restrict__ weights, int* __restrict__ is_sel,
                   int* __restrict__ sel_idx) {
    int i = threadIdx.x;                 // 0..255
    int b = i >> 4, t = i & 15;
    const float* w = weights + b * TT;
    float wt = w[t];
    int rank = 0;
    for (int u = 0; u < TT; ++u) {
        float wu = w[u];
        if (wu > wt || (wu == wt && u < t)) rank++;
    }
    int sel = (rank < KK) ? 1 : 0;
    is_sel[i] = sel;
    if (sel) {
        int slot = 0;
        for (int u = 0; u < t; ++u) {
            float wu = w[u];
            int r2 = 0;
            for (int v = 0; v < TT; ++v) {
                float wv = w[v];
                if (wv > wu || (wv == wu && v < u)) r2++;
            }
            if (r2 < KK) slot++;
        }
        sel_idx[b * KK + slot] = t;      // ascending t -> sorted indices
    }
}

// ---------------- kernel 3: weight transform f32->f16, [tap][co][ci] -------
__global__ __launch_bounds__(256)
void wtrans_kernel(const float* __restrict__ wconv, _Float16* __restrict__ wA) {
    int i = blockIdx.x * 256 + threadIdx.x;
    if (i >= 9 * CC * CC) return;
    int ci  = i & 127;
    int co  = (i >> 7) & 127;
    int tap = i >> 14;
    wA[i] = (_Float16)wconv[((size_t)(co * CC + ci)) * 9 + tap];
}

// ---------------- kernel 4: pass-through copy of non-selected timesteps ----
__global__ __launch_bounds__(256)
void copy_kernel(const float* __restrict__ x, const int* __restrict__ is_sel,
                 float* __restrict__ out) {
    int bt = blockIdx.x;
    if (is_sel[bt]) return;              // conv kernel fills selected slots
    const float4* src = (const float4*)(x   + (size_t)bt * CHW);
    float4*       dst = (float4*)      (out + (size_t)bt * CHW);
    for (int i = threadIdx.x; i < CHW/4; i += 256) dst[i] = src[i];
}

// ---------------- kernel 5: WMMA implicit-GEMM 3x3 conv --------------------
// grid = 128 images * 4 row-blocks; block = 512 threads = 16 waves
// wave w: ct2 = w>>2 -> co_tiles {2*ct2, 2*ct2+1}; nq = w&3 -> n-tiles nq*4..nq*4+3
#define LROWS 10
#define LCOLS 34
#define LCSTR 40    // cell stride in halfs: 80B = 16B aligned, 20-DWORD bank stride
#define SUBC  16    // ci per async sub-stage
__global__ __launch_bounds__(512)
void conv_wmma_kernel(const float* __restrict__ x, const _Float16* __restrict__ wA,
                      const float* __restrict__ bconv, const int* __restrict__ sel_idx,
                      float* __restrict__ out) {
    __shared__ __align__(16) _Float16 tile[LROWS * LCOLS * LCSTR];   // 27,200 B
#if HAS_ASYNC_LDS
    __shared__ __align__(16) float sbuf[SUBC * LROWS * WW];          // 20,480 B
#endif

    int bid = blockIdx.x;
    int img = bid >> 2;                  // 0..127
    int rb  = bid & 3;                   // row block: 8 rows each
    int b   = img >> 3, k = img & 7;
    int t   = sel_idx[b * KK + k];
    const float* ximg = x   + (size_t)(b * TT + t) * CHW;
    float*       oimg = out + (size_t)(b * TT + t) * CHW;
    int h0 = rb * 8;

    int tid     = threadIdx.x;
    int wave    = tid >> 5;
    int lane    = tid & 31;
    int lane_lo = lane & 15;
    int lane_hi = lane >> 4;
    int ct2     = wave >> 2;             // 0..3 -> co tiles {2*ct2, 2*ct2+1}
    int nq      = wave & 3;              // 0..3 -> n tiles nq*4 .. nq*4+3

    v8f acc[2][4] = {};

    for (int ciblk = 0; ciblk < 4; ++ciblk) {
#if HAS_ASYNC_LDS
        // ---- async DMA fp32 rows into LDS, then convert to padded f16 tile ----
        for (int sub = 0; sub < 2; ++sub) {
            int cibase = ciblk * 32 + sub * SUBC;
            // 16 ci * 10 rows * 8 segs of 16B
            for (int u = tid; u < SUBC * LROWS * 8; u += 512) {
                int seg = u & 7;
                int rc  = u >> 3;
                int row = rc % LROWS;
                int ci  = rc / LROWS;
                int h = h0 - 1 + row;
                if ((unsigned)h < (unsigned)HH) {
                    const float* src = ximg + (size_t)(cibase + ci) * (HH*WW) + h * WW + seg * 4;
                    float* dst = &sbuf[(ci * LROWS + row) * WW + seg * 4];
                    __builtin_amdgcn_global_load_async_to_lds_b128(
                        (AS_GLOBAL v4i*)src, (AS_LDS v4i*)dst, 0, 0);
                }
            }
            wait_async0();
            __syncthreads();
            // convert fp32 staging -> f16 channels-last tile (zero halo)
            for (int idx = tid; idx < LROWS * LCOLS * SUBC; idx += 512) {
                int ci  = idx / (LROWS * LCOLS);
                int sp  = idx - ci * (LROWS * LCOLS);
                int row = sp / LCOLS;
                int col = sp - row * LCOLS;
                int h = h0 - 1 + row;
                int w = col - 1;
                float v = 0.f;
                if ((unsigned)h < (unsigned)HH && (unsigned)w < (unsigned)WW)
                    v = sbuf[(ci * LROWS + row) * WW + w];
                tile[(row * LCOLS + col) * LCSTR + sub * SUBC + ci] = (_Float16)v;
            }
            __syncthreads();
        }
#else
        // ---- direct staging: global load + cvt + ds_store ----
        for (int idx = tid; idx < LROWS * LCOLS * 32; idx += 512) {
            int ci  = idx / (LROWS * LCOLS);
            int sp  = idx - ci * (LROWS * LCOLS);
            int row = sp / LCOLS;
            int col = sp - row * LCOLS;
            int h = h0 - 1 + row;
            int w = col - 1;
            float v = 0.f;
            if ((unsigned)h < (unsigned)HH && (unsigned)w < (unsigned)WW)
                v = ximg[(size_t)(ciblk * 32 + ci) * (HH*WW) + h * WW + w];
            tile[(row * LCOLS + col) * LCSTR + ci] = (_Float16)v;
        }
        __syncthreads();
#endif

        // prefetch next ci-block's 128KB while WMMAs run
        if (ciblk < 3) {
            const float* nxt = ximg + (size_t)(ciblk + 1) * 32 * (HH*WW);
            for (int l = tid; l < 1024; l += 512)
                __builtin_prefetch(nxt + l * 32, 0, 0);
        }

        for (int tap = 0; tap < 9; ++tap) {
            int dh = tap / 3, dw = tap - dh * 3;
            // A fragments for the two co tiles
            v16h af[2];
            #pragma unroll
            for (int c = 0; c < 2; ++c) {
                int co_t = ct2 * 2 + c;
                const v8h* ap = (const v8h*)(wA
                    + ((size_t)(tap * CC + co_t * 16 + lane_lo)) * CC
                    + ciblk * 32 + lane_hi * 8);
                af[c] = cat8(ap[0], ap[2]);
            }
            #pragma unroll
            for (int n = 0; n < 4; ++n) {
                int nt  = nq * 4 + n;
                int r   = nt >> 1;
                int ch  = nt & 1;
                int row = r + dh;                       // 0..9
                int col = ch * 16 + lane_lo + dw;       // 0..33
                const v8h* bp = (const v8h*)&tile[(row * LCOLS + col) * LCSTR + lane_hi * 8];
                v16h bf = cat8(bp[0], bp[2]);
                #pragma unroll
                for (int c = 0; c < 2; ++c)
                    acc[c][n] = __builtin_amdgcn_wmma_f32_16x16x32_f16(
                        false, af[c], false, bf, (short)0, acc[c][n], false, false);
            }
        }
        __syncthreads();
    }

    // ---- epilogue: bias + scatter D fragments ----
    float bias[2][8];
    #pragma unroll
    for (int c = 0; c < 2; ++c)
        #pragma unroll
        for (int r = 0; r < 8; ++r)
            bias[c][r] = bconv[(ct2 * 2 + c) * 16 + lane_hi * 8 + r];

    #pragma unroll
    for (int c = 0; c < 2; ++c) {
        #pragma unroll
        for (int n = 0; n < 4; ++n) {
            int nt   = nq * 4 + n;
            int prow = h0 + (nt >> 1);
            int pcol = (nt & 1) * 16 + lane_lo;
            #pragma unroll
            for (int r = 0; r < 8; ++r) {
                int co = (ct2 * 2 + c) * 16 + lane_hi * 8 + r;
                oimg[(size_t)co * (HH*WW) + prow * WW + pcol] = acc[c][n][r] + bias[c][r];
            }
        }
    }
}

// ---------------- launcher ----------------
extern "C" void kernel_launch(void* const* d_in, const int* in_sizes, int n_in,
                              void* d_out, int out_size, void* d_ws, size_t ws_size,
                              hipStream_t stream) {
    const float* x        = (const float*)d_in[0];
    const float* w_router = (const float*)d_in[1];
    const float* b_router = (const float*)d_in[2];
    const float* w_conv   = (const float*)d_in[3];
    const float* b_conv   = (const float*)d_in[4];
    float* out = (float*)d_out;

    char* ws = (char*)d_ws;
    float*    weights = (float*)ws;                 // 256 f32
    int*      is_sel  = (int*)(ws + 1024);          // 256 i32
    int*      sel_idx = (int*)(ws + 2048);          // 128 i32
    _Float16* wA      = (_Float16*)(ws + 4096);     // 9*128*128 f16 = 288KB

    router_kernel<<<BB*TT, 256, 0, stream>>>(x, w_router, b_router, weights);
    select_kernel<<<1, 256, 0, stream>>>(weights, is_sel, sel_idx);
    wtrans_kernel<<<(9*CC*CC + 255)/256, 256, 0, stream>>>(w_conv, wA);
    copy_kernel<<<BB*TT, 256, 0, stream>>>(x, is_sel, out);
    conv_wmma_kernel<<<128*4, 512, 0, stream>>>(x, wA, b_conv, sel_idx, out);
}
